// TransformerBlock_14688788152483
// MI455X (gfx1250) — compile-verified, matrix operand order checked by reading, not run on hardware
//
#include <hip/hip_runtime.h>

// ---------------------------------------------------------------------------
// MI455X (gfx1250, wave32) transformer block.
// All matmuls: v_wmma_f32_16x16x32_f16 (f16 operands, f32 accumulate).
// Global->LDS staging uses the CDNA5 async-LDS path when available
// (GLOBAL_LOAD_ASYNC_TO_LDS_B128 + s_wait_asynccnt), double-buffered.
// ---------------------------------------------------------------------------

typedef _Float16 v16h __attribute__((ext_vector_type(16)));
typedef _Float16 v8h  __attribute__((ext_vector_type(8)));
typedef _Float16 v4h  __attribute__((ext_vector_type(4)));
typedef float    v8f  __attribute__((ext_vector_type(8)));
typedef int      v4i  __attribute__((ext_vector_type(4)));

#if defined(__has_builtin)
#  if __has_builtin(__builtin_amdgcn_global_load_async_to_lds_b128) && \
      __has_builtin(__builtin_amdgcn_s_wait_asynccnt)
#    define USE_ASYNC_LDS 1
#  endif
#endif
#ifndef USE_ASYNC_LDS
#  define USE_ASYNC_LDS 0
#endif

#if USE_ASYNC_LDS
// b128 async copy payload is typed as int4; global src is AS1, LDS dst is AS3.
typedef __attribute__((address_space(1))) v4i gbl_v4i;
typedef __attribute__((address_space(3))) v4i lds_v4i;
#endif

// Copy 16 bytes global -> LDS (per-lane addresses).
__device__ __forceinline__ void cp16(_Float16* lds, const _Float16* g) {
#if USE_ASYNC_LDS
  _Float16* gn = const_cast<_Float16*>(g);
  __builtin_amdgcn_global_load_async_to_lds_b128((gbl_v4i*)gn, (lds_v4i*)lds,
                                                 0, 0);
#else
  *(v8h*)lds = *(const v8h*)g;
#endif
}
__device__ __forceinline__ void async_fence() {
#if USE_ASYNC_LDS
  __builtin_amdgcn_s_wait_asynccnt(0);
#endif
}

__device__ __forceinline__ v16h mk16(v8h a, v8h b) {
  return __builtin_shufflevector(a, b, 0,1,2,3,4,5,6,7,8,9,10,11,12,13,14,15);
}
// Load an A/B fragment: 8 halves at p (K = kk0..kk0+7) and 8 at p+16
// (K = kk0+16..kk0+23), matching the CDNA5 16-bit A-matrix VGPR layout.
__device__ __forceinline__ v16h frag_ld(const _Float16* p) {
  v8h a = *(const v8h*)(p);
  v8h b = *(const v8h*)(p + 16);
  return mk16(a, b);
}
__device__ __forceinline__ v8f wmma16(v16h a, v16h b, v8f c) {
  return __builtin_amdgcn_wmma_f32_16x16x32_f16(false, a, false, b, (short)0, c,
                                                false, false);
}

// ---------------------------------------------------------------------------
// LayerNorm (D = 1024), f32 in -> f16 out. One 256-thread block per row.
// ---------------------------------------------------------------------------
__global__ __launch_bounds__(256) void ln_f16_kernel(
    const float* __restrict__ x, const float* __restrict__ gamma,
    const float* __restrict__ beta, _Float16* __restrict__ out) {
  const int D = 1024;
  int row = blockIdx.x, tid = threadIdx.x;
  float4 v = ((const float4*)(x + (size_t)row * D))[tid];
  float s = v.x + v.y + v.z + v.w;
  float q = v.x * v.x + v.y * v.y + v.z * v.z + v.w * v.w;
  __shared__ float r1[256], r2[256];
  r1[tid] = s; r2[tid] = q;
  __syncthreads();
  for (int st = 128; st > 0; st >>= 1) {
    if (tid < st) { r1[tid] += r1[tid + st]; r2[tid] += r2[tid + st]; }
    __syncthreads();
  }
  float mu  = r1[0] * (1.0f / 1024.0f);
  float var = r2[0] * (1.0f / 1024.0f) - mu * mu;
  float rs  = rsqrtf(var + 1e-5f);
  float4 g = ((const float4*)gamma)[tid];
  float4 b = ((const float4*)beta)[tid];
  v4h o;
  o[0] = (_Float16)((v.x - mu) * rs * g.x + b.x);
  o[1] = (_Float16)((v.y - mu) * rs * g.y + b.y);
  o[2] = (_Float16)((v.z - mu) * rs * g.z + b.z);
  o[3] = (_Float16)((v.w - mu) * rs * g.w + b.w);
  ((v4h*)(out + (size_t)row * D))[tid] = o;
}

// ---------------------------------------------------------------------------
// Weight prep: f32 [R][C] -> f16 [C][R] (transpose + cast), 32x32 LDS tiles.
// ---------------------------------------------------------------------------
__global__ __launch_bounds__(256) void transpose_cast_kernel(
    const float* __restrict__ in, _Float16* __restrict__ out,
    int R, int C, long long inBatch, long long outBatch) {
  in  += (long long)blockIdx.z * inBatch;
  out += (long long)blockIdx.z * outBatch;
  int r0 = blockIdx.y * 32, c0 = blockIdx.x * 32;
  __shared__ float t[32][33];
  int tid = threadIdx.x;
  for (int i = 0; i < 4; ++i) {
    int e = i * 256 + tid, r = e >> 5, c = e & 31;
    t[r][c] = in[(size_t)(r0 + r) * C + (c0 + c)];
  }
  __syncthreads();
  for (int i = 0; i < 4; ++i) {
    int e = i * 256 + tid, r = e >> 5, c = e & 31;
    out[(size_t)(c0 + r) * R + (r0 + c)] = (_Float16)t[c][r];
  }
}

// ---------------------------------------------------------------------------
// Generic WMMA GEMM: C[M,N] = act(A[M,K] * B^T[N,K] + bias) (+ residual).
// 256 threads = 8 waves; block tile 128x128; wave tile 32x64; K-step 32.
// Double-buffered LDS (one barrier / K-step); async global->LDS staging.
// LDS row stride 40 halves (80B = 20 banks -> conflict-free b128 reads).
// ---------------------------------------------------------------------------
__global__ __launch_bounds__(256) void gemm_f16_kernel(
    const _Float16* __restrict__ A, const _Float16* __restrict__ BT,
    const float* __restrict__ bias, const float* __restrict__ resid,
    float* __restrict__ outF, _Float16* __restrict__ outH,
    int M, int N, int K, int doRelu) {
  const int LS = 40;
  __shared__ _Float16 Al[2][128 * LS];
  __shared__ _Float16 Bl[2][128 * LS];
  int tid  = threadIdx.x;
  int wave = tid >> 5, lane = tid & 31;
  int hi = lane >> 4, lo = lane & 15, kk0 = hi * 8;
  int wm = wave >> 1, wn = wave & 1;
  int m0 = blockIdx.y * 128, n0 = blockIdx.x * 128;
  int ldr = tid >> 1, ldc = (tid & 1) * 16;
  int dOff = ldr * LS + ldc;
  const _Float16* Ag = A  + (size_t)(m0 + ldr) * K + ldc;
  const _Float16* Bg = BT + (size_t)(n0 + ldr) * K + ldc;

  auto stage = [&](int buf, int k0) {
    cp16(&Al[buf][dOff],     Ag + k0);
    cp16(&Al[buf][dOff + 8], Ag + k0 + 8);
    cp16(&Bl[buf][dOff],     Bg + k0);
    cp16(&Bl[buf][dOff + 8], Bg + k0 + 8);
  };

  v8f acc[2][4] = {};
  int nk = K >> 5;
  stage(0, 0);
  for (int i = 0; i < nk; ++i) {
    int cur = i & 1;
    async_fence();           // this wave's async writes into LDS are done
    __syncthreads();         // everyone's writes visible; prev reads done
    if (i + 1 < nk) stage(cur ^ 1, (i + 1) << 5);
    if (i + 2 < nk) {
      __builtin_prefetch(Ag + ((i + 2) << 5), 0, 1);  // global_prefetch_b8
      __builtin_prefetch(Bg + ((i + 2) << 5), 0, 1);
    }
    const _Float16* Ab = Al[cur];
    const _Float16* Bb = Bl[cur];
    v16h af0 = frag_ld(&Ab[(wm * 32 + lo) * LS + kk0]);
    v16h af1 = frag_ld(&Ab[(wm * 32 + 16 + lo) * LS + kk0]);
    v16h bf[4];
    for (int nt = 0; nt < 4; ++nt)
      bf[nt] = frag_ld(&Bb[(wn * 64 + nt * 16 + lo) * LS + kk0]);
    for (int nt = 0; nt < 4; ++nt) {
      acc[0][nt] = wmma16(af0, bf[nt], acc[0][nt]);
      acc[1][nt] = wmma16(af1, bf[nt], acc[1][nt]);
    }
  }
  for (int gm = 0; gm < 2; ++gm)
    for (int nt = 0; nt < 4; ++nt) {
      int col = n0 + wn * 64 + nt * 16 + lo;
      float bv = bias ? bias[col] : 0.0f;
      for (int g = 0; g < 8; ++g) {
        int m = m0 + wm * 32 + gm * 16 + hi * 8 + g;
        float v = acc[gm][nt][g] + bv;
        if (doRelu) v = fmaxf(v, 0.0f);
        size_t off = (size_t)m * N + col;
        if (resid) v += resid[off];
        if (outF) outF[off] = v;
        if (outH) outH[off] = (_Float16)v;
      }
    }
}

// ---------------------------------------------------------------------------
// Flash attention, causal. Block = 8 waves x 16 queries = 128 queries of one
// (b, h). K-tile (async) / V^T-tile staged in LDS (stride 72, conflict-free).
// Online softmax per row via 16-lane shfl reductions (wave32 halves).
// P converts C-layout -> A-layout through wave-private LDS.
// ---------------------------------------------------------------------------
__global__ __launch_bounds__(256) void flash_attn_kernel(
    const _Float16* __restrict__ qkv, _Float16* __restrict__ outH) {
  const int T = 2048, S3 = 3072, HD = 64;
  int qb = blockIdx.x, h = blockIdx.y, b = blockIdx.z;
  int tid  = threadIdx.x;
  int wave = tid >> 5, lane = tid & 31;
  int hi = lane >> 4, lo = lane & 15, kk0 = hi * 8;
  int q0 = qb * 128 + wave * 16;

  const _Float16* Qp = qkv + (size_t)(b * T) * S3 + h * HD;
  const _Float16* Kp = Qp + 1024;
  const _Float16* Vp = Qp + 2048;

  __shared__ _Float16 Ksh[64 * 72];
  __shared__ _Float16 VTsh[64 * 72];
  __shared__ _Float16 Psh[8 * 16 * 72];

  v16h qf0, qf1;
  {
    const _Float16* qrow = Qp + (size_t)(q0 + lo) * S3;
    qf0 = mk16(*(const v8h*)(qrow + kk0),      *(const v8h*)(qrow + kk0 + 16));
    qf1 = mk16(*(const v8h*)(qrow + 32 + kk0), *(const v8h*)(qrow + 32 + kk0 + 16));
  }

  v8f Oacc[4] = {};
  float rowM[8], rowL[8];
  for (int g = 0; g < 8; ++g) { rowM[g] = -1e30f; rowL[g] = 0.0f; }

  _Float16* Pw = &Psh[wave * 16 * 72];
  int ldr = tid >> 2, cg = tid & 3;
  int ntiles = 2 * (qb + 1);  // keys up to end of this query block

  for (int kt = 0; kt < ntiles; ++kt) {
    int k0 = kt * 64;
    __syncthreads();
    {  // K tile: straight per-lane copy -> async path; V: transposed (sync)
      const _Float16* krow = Kp + (size_t)(k0 + ldr) * S3 + cg * 16;
      cp16(&Ksh[ldr * 72 + cg * 16], krow);
      cp16(&Ksh[ldr * 72 + cg * 16 + 8], krow + 8);
      const _Float16* vrow = Vp + (size_t)(k0 + ldr) * S3 + cg * 16;
      v8h v0 = *(const v8h*)(vrow);
      v8h v1 = *(const v8h*)(vrow + 8);
      for (int j = 0; j < 8; ++j) {
        VTsh[(cg * 16 + j) * 72 + ldr]     = v0[j];
        VTsh[(cg * 16 + 8 + j) * 72 + ldr] = v1[j];
      }
    }
    async_fence();
    __syncthreads();

    if (k0 <= q0 + 15) {  // uniform per wave
      float p[4][8];
      for (int nt = 0; nt < 4; ++nt) {
        v8f s = {};
        v16h bf0 = frag_ld(&Ksh[(nt * 16 + lo) * 72 + kk0]);
        v16h bf1 = frag_ld(&Ksh[(nt * 16 + lo) * 72 + 32 + kk0]);
        s = wmma16(qf0, bf0, s);
        s = wmma16(qf1, bf1, s);
        for (int g = 0; g < 8; ++g) {
          int qi = q0 + hi * 8 + g;
          int kj = k0 + nt * 16 + lo;
          p[nt][g] = (kj <= qi) ? s[g] * 0.125f : -1e30f;
        }
      }
      for (int g = 0; g < 8; ++g) {
        float m = fmaxf(fmaxf(p[0][g], p[1][g]), fmaxf(p[2][g], p[3][g]));
        for (int off = 1; off < 16; off <<= 1)
          m = fmaxf(m, __shfl_xor(m, off, 32));
        float nm = fmaxf(rowM[g], m);
        float corr = __expf(rowM[g] - nm);
        float rsum = 0.0f;
        for (int nt = 0; nt < 4; ++nt) {
          float e = __expf(p[nt][g] - nm);
          p[nt][g] = e;
          rsum += e;
        }
        for (int off = 1; off < 16; off <<= 1)
          rsum += __shfl_xor(rsum, off, 32);
        rowL[g] = rowL[g] * corr + rsum;
        rowM[g] = nm;
        for (int et = 0; et < 4; ++et) Oacc[et][g] *= corr;
      }
      // P: C-fragment layout -> wave-private LDS -> A-fragment layout
      for (int nt = 0; nt < 4; ++nt)
        for (int g = 0; g < 8; ++g)
          Pw[(hi * 8 + g) * 72 + nt * 16 + lo] = (_Float16)p[nt][g];
      v16h pf0 = frag_ld(&Pw[lo * 72 + kk0]);
      v16h pf1 = frag_ld(&Pw[lo * 72 + 32 + kk0]);
      for (int et = 0; et < 4; ++et) {
        v16h vf0 = frag_ld(&VTsh[(et * 16 + lo) * 72 + kk0]);
        v16h vf1 = frag_ld(&VTsh[(et * 16 + lo) * 72 + 32 + kk0]);
        Oacc[et] = wmma16(pf0, vf0, Oacc[et]);
        Oacc[et] = wmma16(pf1, vf1, Oacc[et]);
      }
    }
  }

  for (int g = 0; g < 8; ++g) {
    float inv = 1.0f / rowL[g];
    size_t orow = (size_t)(b * T + q0 + hi * 8 + g) * 1024 + h * HD;
    for (int et = 0; et < 4; ++et)
      outH[orow + et * 16 + lo] = (_Float16)(Oacc[et][g] * inv);
  }
}

// ---------------------------------------------------------------------------
extern "C" void kernel_launch(void* const* d_in, const int* in_sizes, int n_in,
                              void* d_out, int out_size, void* d_ws, size_t ws_size,
                              hipStream_t stream) {
  (void)in_sizes; (void)n_in; (void)out_size; (void)ws_size;
  const int B = 4, T = 2048, D = 1024, H = 16, HD = 64, F = 4096;
  const int M = B * T;

  const float* x   = (const float*)d_in[0];
  const float* Wq  = (const float*)d_in[1];
  const float* Wk  = (const float*)d_in[2];
  const float* Wv  = (const float*)d_in[3];
  const float* Wo  = (const float*)d_in[4];
  const float* bo  = (const float*)d_in[5];
  const float* W1  = (const float*)d_in[6];
  const float* b1  = (const float*)d_in[7];
  const float* W2  = (const float*)d_in[8];
  const float* b2  = (const float*)d_in[9];
  const float* g1  = (const float*)d_in[10];
  const float* be1 = (const float*)d_in[11];
  const float* g2  = (const float*)d_in[12];
  const float* be2 = (const float*)d_in[13];

  char* w = (char*)d_ws;
  auto alloc = [&](size_t bytes) -> void* {
    void* p = (void*)w;
    w += (bytes + 255) & ~(size_t)255;
    return p;
  };
  _Float16* h16  = (_Float16*)alloc((size_t)M * D * 2);       // LN1 out
  _Float16* qkvT = (_Float16*)alloc((size_t)3 * D * D * 2);   // [3072][1024]
  _Float16* woT  = (_Float16*)alloc((size_t)D * D * 2);       // [1024][1024]
  _Float16* w1T  = (_Float16*)alloc((size_t)F * D * 2);       // [4096][1024]
  _Float16* w2T  = (_Float16*)alloc((size_t)D * F * 2);       // [1024][4096]
  _Float16* qkv  = (_Float16*)alloc((size_t)M * 3 * D * 2);   // [8192][3072]
  _Float16* a16  = (_Float16*)alloc((size_t)M * D * 2);       // attn concat
  float*    x1   = (float*)   alloc((size_t)M * D * 4);       // residual 1
  _Float16* h2   = (_Float16*)alloc((size_t)M * D * 2);       // LN2 out
  _Float16* ffh  = (_Float16*)alloc((size_t)M * F * 2);       // relu(h2 W1+b1)

  // Weight prep: cast + transpose so every GEMM B operand is B^T[N][K].
  transpose_cast_kernel<<<dim3(HD / 32, D / 32, H), 256, 0, stream>>>(
      Wq, qkvT, D, HD, (long long)D * HD, (long long)HD * D);
  transpose_cast_kernel<<<dim3(HD / 32, D / 32, H), 256, 0, stream>>>(
      Wk, qkvT + (size_t)D * D, D, HD, (long long)D * HD, (long long)HD * D);
  transpose_cast_kernel<<<dim3(HD / 32, D / 32, H), 256, 0, stream>>>(
      Wv, qkvT + (size_t)2 * D * D, D, HD, (long long)D * HD, (long long)HD * D);
  transpose_cast_kernel<<<dim3(D / 32, D / 32, 1), 256, 0, stream>>>(
      Wo, woT, D, D, 0, 0);
  transpose_cast_kernel<<<dim3(F / 32, D / 32, 1), 256, 0, stream>>>(
      W1, w1T, D, F, 0, 0);
  transpose_cast_kernel<<<dim3(D / 32, F / 32, 1), 256, 0, stream>>>(
      W2, w2T, F, D, 0, 0);

  // LN1 -> fused QKV projection -> flash attention
  ln_f16_kernel<<<M, 256, 0, stream>>>(x, g1, be1, h16);
  gemm_f16_kernel<<<dim3(3 * D / 128, M / 128), 256, 0, stream>>>(
      h16, qkvT, nullptr, nullptr, nullptr, qkv, M, 3 * D, D, 0);
  flash_attn_kernel<<<dim3(T / 128, H, B), 256, 0, stream>>>(qkv, a16);

  // Wo projection + bias + residual(x) -> x1 (f32)
  gemm_f16_kernel<<<dim3(D / 128, M / 128), 256, 0, stream>>>(
      a16, woT, bo, x, x1, nullptr, M, D, D, 0);

  // LN2 -> FFN1 (relu) -> FFN2 + residual(x1) -> out
  ln_f16_kernel<<<M, 256, 0, stream>>>(x1, g2, be2, h2);
  gemm_f16_kernel<<<dim3(F / 128, M / 128), 256, 0, stream>>>(
      h2, w1T, b1, nullptr, nullptr, ffh, M, F, D, 1);
  gemm_f16_kernel<<<dim3(D / 128, M / 128), 256, 0, stream>>>(
      ffh, w2T, b2, x1, (float*)d_out, nullptr, M, D, F, 0);
}
